// AttentionHead_17437567222165
// MI455X (gfx1250) — matrix operand order repaired and checked by reading, not compiled, not run on hardware
//
#include <hip/hip_runtime.h>
#include <hip/hip_bf16.h>

typedef __attribute__((ext_vector_type(16))) _Float16 v16h;
typedef __attribute__((ext_vector_type(8)))  _Float16 v8h;
typedef __attribute__((ext_vector_type(8)))  float    v8f;
typedef __attribute__((ext_vector_type(4)))  float    v4f;

namespace {
constexpr int kB = 4;      // batch
constexpr int kT = 4096;   // sequence length
constexpr int kC = 512;    // model dim
constexpr int kH = 64;     // head size
}

// D = A(16x32 f16) * B(32x16 f16) + C(16x16 f32)
static __device__ inline v8f wmma16(v16h a, v16h b, v8f c) {
  return __builtin_amdgcn_wmma_f32_16x16x32_f16(
      /*neg_a=*/false, a, /*neg_b=*/false, b,
      /*c_mod=*/(short)0, c, /*reuse_a=*/false, /*reuse_b=*/false);
}

// A-fragment (16x32, f16): lane holds its row; K halves [0..7] at p, [16..23] at p+16
// (caller adds +8 for upper lane half). p must be 16B aligned.
static __device__ inline v16h loadA16(const _Float16* p) {
  v8h lo = *reinterpret_cast<const v8h*>(p);
  v8h hh = *reinterpret_cast<const v8h*>(p + 16);
  v16h r;
#pragma unroll
  for (int i = 0; i < 8; ++i) { r[i] = lo[i]; r[i + 8] = hh[i]; }
  return r;
}

// B-fragment (32x16, f16): lane holds 16 contiguous K values (caller adds +16 for
// upper lane half). p must be 16B aligned.
static __device__ inline v16h loadB16(const _Float16* p) {
  v8h lo = *reinterpret_cast<const v8h*>(p);
  v8h hh = *reinterpret_cast<const v8h*>(p + 8);
  v16h r;
#pragma unroll
  for (int i = 0; i < 8; ++i) { r[i] = lo[i]; r[i + 8] = hh[i]; }
  return r;
}

// A-fragment built from fp32 memory (convert to f16 on the fly).
static __device__ inline v16h loadA_f32(const float* p) {
  v4f a = *reinterpret_cast<const v4f*>(p);
  v4f b = *reinterpret_cast<const v4f*>(p + 4);
  v4f c = *reinterpret_cast<const v4f*>(p + 16);
  v4f d = *reinterpret_cast<const v4f*>(p + 20);
  v16h r;
#pragma unroll
  for (int i = 0; i < 4; ++i) {
    r[i]      = (_Float16)a[i];
    r[i + 4]  = (_Float16)b[i];
    r[i + 8]  = (_Float16)c[i];
    r[i + 12] = (_Float16)d[i];
  }
  return r;
}

// ---------------------------------------------------------------------------
// Kernel 0: Wt[m][h][c] = (f16) W_m[c][h]   (m = q,k,v)  -> contiguous B-frag loads
// ---------------------------------------------------------------------------
__global__ __launch_bounds__(256) void prep_wt_kernel(
    const float* __restrict__ Wq, const float* __restrict__ Wk,
    const float* __restrict__ Wv, _Float16* __restrict__ Wt) {
  const int per = kH * kC;
  int idx = blockIdx.x * 256 + threadIdx.x;
  if (idx >= 3 * per) return;
  int m = idx / per;
  int r = idx - m * per;
  int h = r / kC;
  int c = r - h * kC;
  const float* W = (m == 0) ? Wq : (m == 1) ? Wk : Wv;
  Wt[idx] = (_Float16)W[c * kH + h];
}

// ---------------------------------------------------------------------------
// Kernel 1: QKV projection. One wave per 16-row tile of flattened [B*T, C] x.
// q is pre-scaled by 1/sqrt(H). Q,K stored row-major f16 [B*T][64];
// V stored transposed f16 Vt[B][64][T] so PV A-fragments are contiguous.
// ---------------------------------------------------------------------------
__global__ __launch_bounds__(256) void qkv_kernel(
    const float* __restrict__ x, const _Float16* __restrict__ Wt,
    const float* __restrict__ bq, const float* __restrict__ bk,
    const float* __restrict__ bv,
    _Float16* __restrict__ Qh, _Float16* __restrict__ Kh,
    _Float16* __restrict__ Vt) {
  const int lane = threadIdx.x & 31;
  const int wid  = threadIdx.x >> 5;
  const int tile = blockIdx.x * 8 + wid;       // 16-row tile of [B*T]
  const int n    = lane & 15;
  const int hi   = lane >> 4;
  const long r0  = (long)tile * 16;

  const float* xrow = x + (r0 + n) * kC;       // A: row = lane%16

  v8f acc[3][4] = {};                          // 3 matrices x 4 N-tiles

  for (int ks = 0; ks < kC / 32; ++ks) {       // 16 K-steps of 32
    v16h a = loadA_f32(xrow + ks * 32 + hi * 8);
#pragma unroll
    for (int m = 0; m < 3; ++m) {
      const _Float16* wm = Wt + m * (kH * kC);
#pragma unroll
      for (int t = 0; t < 4; ++t) {
        v16h bf = loadA16(wm + (t * 16 + n) * kC + ks * 32 + hi * 8);
        acc[m][t] = wmma16(a, bf, acc[m][t]);
      }
    }
  }

  const int bidx = (int)(r0 / kT);
  const int t0   = (int)(r0 - (long)bidx * kT);
#pragma unroll
  for (int t = 0; t < 4; ++t) {
    const int col = t * 16 + n;                // C/D: N = lane%16
    const float aq = bq[col];
    const float ak = bk[col];
    const float av = bv[col];
#pragma unroll
    for (int v = 0; v < 8; ++v) {              // C/D: row = VGPR idx (+8 upper half)
      const long row = r0 + v + hi * 8;
      Qh[row * kH + col] = (_Float16)((acc[0][t][v] + aq) * 0.125f); // fold 1/sqrt(64)
      Kh[row * kH + col] = (_Float16)(acc[1][t][v] + ak);
      Vt[((long)bidx * kH + col) * kT + (t0 + v + hi * 8)] =
          (_Float16)(acc[2][t][v] + av);
    }
  }
}

// ---------------------------------------------------------------------------
// Masking + online-softmax update for one 16-query group over a 32-key step.
// s1/s2 are S^T tiles (keys x queries, C-layout). Updates (m, l, o[4]) and
// returns P^T packed into the f16 B-fragment layout for the PV WMMAs.
// ---------------------------------------------------------------------------
static __device__ __forceinline__ v16h mask_softmax_pb(
    v8f s1, v8f s2, unsigned long long pm1, unsigned long long pm2,
    int k0, int hi, int qg, float& m, float& l, v8f o[4]) {
#pragma unroll
  for (int v = 0; v < 8; ++v) {                // causal + padding: overwrite S
    const int kg1 = k0 + hi * 8 + v;
    const int kg2 = kg1 + 16;
    const bool ok1 = (kg1 <= qg) & (((pm1 >> (8 * v)) & 0xffull) == 0);
    const bool ok2 = (kg2 <= qg) & (((pm2 >> (8 * v)) & 0xffull) == 0);
    s1[v] = ok1 ? s1[v] : -1e38f;
    s2[v] = ok2 ? s2[v] : -1e38f;
  }
  float tm = -1e38f;
#pragma unroll
  for (int v = 0; v < 8; ++v) tm = fmaxf(tm, fmaxf(s1[v], s2[v]));
  tm = fmaxf(tm, __shfl_xor(tm, 16, 32));      // combine lane halves (same query)
  const float nm    = fmaxf(m, tm);
  const float alpha = __expf(m - nm);
  m = nm;

  float e1[8], e2[8];
  float ts = 0.f;
#pragma unroll
  for (int v = 0; v < 8; ++v) {
    e1[v] = __expf(s1[v] - nm);
    e2[v] = __expf(s2[v] - nm);
    ts += e1[v] + e2[v];
  }
  ts += __shfl_xor(ts, 16, 32);
  l = l * alpha + ts;

#pragma unroll
  for (int t = 0; t < 4; ++t) o[t] *= alpha;

  // Re-lay P^T (C-layout f32) into B-fragment f16: lanes<16 need keys 0..15,
  // lanes>=16 need keys 16..31; half of each comes from the opposite lane half.
  float sx1[8], sx2[8];
#pragma unroll
  for (int v = 0; v < 8; ++v) {
    sx1[v] = __shfl_xor(e1[v], 16, 32);
    sx2[v] = __shfl_xor(e2[v], 16, 32);
  }
  v16h pb;
#pragma unroll
  for (int j = 0; j < 8; ++j) {
    pb[j]     = (_Float16)(hi ? sx2[j] : e1[j]);   // keys j   / 16+j
    pb[j + 8] = (_Float16)(hi ? e2[j]  : sx1[j]);  // keys 8+j / 24+j
  }
  return pb;
}

// ---------------------------------------------------------------------------
// Kernel 2: causal flash attention, S^T = K*Q^T orientation. One wave handles
// 32 queries (two 16-query B-fragments) so every K and V A-fragment load is
// reused by two query groups (2x arithmetic intensity vs one group). Per
// 32-key step: 8 WMMAs for S^T, two online-softmax updates, 8 WMMAs for
// O^T += V^T * P^T. Longest (largest q0) tiles are launched first to reduce
// causal load-imbalance tail.
// ---------------------------------------------------------------------------
__global__ __launch_bounds__(256) void flash_attn_kernel(
    const _Float16* __restrict__ Qh, const _Float16* __restrict__ Kh,
    const _Float16* __restrict__ Vt, const unsigned char* __restrict__ pmask,
    float* __restrict__ out) {
  const int lane = threadIdx.x & 31;
  const int wid  = threadIdx.x >> 5;
  const int tile = blockIdx.x * 8 + wid;       // 0 .. B*T/32-1
  const int tilesPerB = kT / 32;               // 128
  const int b  = tile / tilesPerB;
  const int j  = tile - b * tilesPerB;
  const int q0 = (tilesPerB - 1 - j) * 32;     // reversed: long tiles first
  const int n  = lane & 15;
  const int hi = lane >> 4;
  const int qgA = q0 + n;                      // query group A
  const int qgB = q0 + 16 + n;                 // query group B

  // Q as B-matrix fragments (head dim is the contraction): contiguous per lane.
  const _Float16* qbase = Qh + ((long)(b * kT + q0 + n)) * kH + hi * 16;
  const v16h qa0 = loadB16(qbase);             // group A, head 0..31
  const v16h qa1 = loadB16(qbase + 32);        // group A, head 32..63
  const v16h qb0 = loadB16(qbase + 16 * kH);   // group B
  const v16h qb1 = loadB16(qbase + 16 * kH + 32);

  v8f oA[4] = {}, oB[4] = {};                  // O^T, 4 head chunks each
  float mA = -1e30f, lA = 0.f;
  float mB = -1e30f, lB = 0.f;

  const int kend = q0 + 32;                    // causal: keys <= q0+31 (group B)
  for (int k0 = 0; k0 < kend; k0 += 32) {      // max k0 = T-32 -> loads in-bounds
    const _Float16* krow = Kh + ((long)(b * kT + k0 + n)) * kH + hi * 16;
    __builtin_prefetch(krow + 32 * kH, 0, 1);  // next key tile

    v16h ka0 = loadB16(krow);                  // keys k0..k0+15, head 0..31
    v16h ka1 = loadB16(krow + 32);             //                 head 32..63
    v16h kc0 = loadB16(krow + 16 * kH);        // keys k0+16..k0+31
    v16h kc1 = loadB16(krow + 16 * kH + 32);

    v8f sA1 = {};
    sA1 = wmma16(ka0, qa0, sA1);
    sA1 = wmma16(ka1, qa1, sA1);
    v8f sA2 = {};
    sA2 = wmma16(kc0, qa0, sA2);
    sA2 = wmma16(kc1, qa1, sA2);
    v8f sB1 = {};
    sB1 = wmma16(ka0, qb0, sB1);
    sB1 = wmma16(ka1, qb1, sB1);
    v8f sB2 = {};
    sB2 = wmma16(kc0, qb0, sB2);
    sB2 = wmma16(kc1, qb1, sB2);

    // padding mask: 8 consecutive key bytes per lane-half (shared by A and B)
    const unsigned char* pmb = pmask + (long)b * kT + k0 + hi * 8;
    const unsigned long long pm1 = *reinterpret_cast<const unsigned long long*>(pmb);
    const unsigned long long pm2 = *reinterpret_cast<const unsigned long long*>(pmb + 16);

    const v16h pbA = mask_softmax_pb(sA1, sA2, pm1, pm2, k0, hi, qgA, mA, lA, oA);
    const v16h pbB = mask_softmax_pb(sB1, sB2, pm1, pm2, k0, hi, qgB, mB, lB, oB);

    // O^T[h,q] += V^T[h,k] * P^T[k,q]; each V fragment feeds both query groups.
#pragma unroll
    for (int t = 0; t < 4; ++t) {
      const _Float16* vp =
          Vt + ((long)(b * kH + t * 16 + n)) * kT + k0 + hi * 16;
      v16h va = loadB16(vp);
      oA[t] = wmma16(va, pbA, oA[t]);
      oB[t] = wmma16(va, pbB, oB[t]);
    }
  }

  // out[b, q, h] = O^T[h, q] / l   (8 consecutive h per VGPR run -> b128 stores)
  const float rlA = 1.f / lA;
  const float rlB = 1.f / lB;
  float* poA = out + ((long)(b * kT + q0 + n)) * kH + hi * 8;
  float* poB = poA + (long)16 * kH;
#pragma unroll
  for (int t = 0; t < 4; ++t) {
    v4f a0 = { oA[t][0] * rlA, oA[t][1] * rlA, oA[t][2] * rlA, oA[t][3] * rlA };
    v4f a1 = { oA[t][4] * rlA, oA[t][5] * rlA, oA[t][6] * rlA, oA[t][7] * rlA };
    *reinterpret_cast<v4f*>(poA + t * 16)     = a0;
    *reinterpret_cast<v4f*>(poA + t * 16 + 4) = a1;
    v4f b0 = { oB[t][0] * rlB, oB[t][1] * rlB, oB[t][2] * rlB, oB[t][3] * rlB };
    v4f b1 = { oB[t][4] * rlB, oB[t][5] * rlB, oB[t][6] * rlB, oB[t][7] * rlB };
    *reinterpret_cast<v4f*>(poB + t * 16)     = b0;
    *reinterpret_cast<v4f*>(poB + t * 16 + 4) = b1;
  }
}

extern "C" void kernel_launch(void* const* d_in, const int* in_sizes, int n_in,
                              void* d_out, int out_size, void* d_ws, size_t ws_size,
                              hipStream_t stream) {
  (void)in_sizes; (void)n_in; (void)out_size; (void)ws_size;

  const float*         x     = (const float*)d_in[0];
  const unsigned char* pmask = (const unsigned char*)d_in[1];
  const float*         Wq    = (const float*)d_in[2];
  const float*         bq    = (const float*)d_in[3];
  const float*         Wk    = (const float*)d_in[4];
  const float*         bk    = (const float*)d_in[5];
  const float*         Wv    = (const float*)d_in[6];
  const float*         bv    = (const float*)d_in[7];
  float*               out   = (float*)d_out;

  // Workspace layout: Wt (192KB) | Qh (2MB) | Kh (2MB) | Vt (2MB)
  char* ws = (char*)d_ws;
  _Float16* Wt = (_Float16*)ws;
  size_t off = (size_t)3 * kH * kC * sizeof(_Float16);
  _Float16* Qh = (_Float16*)(ws + off); off += (size_t)kB * kT * kH * sizeof(_Float16);
  _Float16* Kh = (_Float16*)(ws + off); off += (size_t)kB * kT * kH * sizeof(_Float16);
  _Float16* Vt = (_Float16*)(ws + off);

  prep_wt_kernel<<<(3 * kH * kC + 255) / 256, 256, 0, stream>>>(Wq, Wk, Wv, Wt);

  const int rowTiles = kB * kT / 16;           // 1024 row-tiles, 8 waves/block
  qkv_kernel<<<rowTiles / 8, 256, 0, stream>>>(x, Wt, bq, bk, bv, Qh, Kh, Vt);

  const int qTiles = kB * kT / 32;             // 512 query-tiles of 32, 8 waves/block
  flash_attn_kernel<<<qTiles / 8, 256, 0, stream>>>(Qh, Kh, Vt, pmask, out);
}
